// RGUnit_34978213658993
// MI455X (gfx1250) — matrix operand order repaired and checked by reading, not compiled, not run on hardware
//
#include <hip/hip_runtime.h>
#include <hip/hip_bf16.h>
#include <math.h>

// ---------------------------------------------------------------------------
// RGU (reciprocally-gated recurrent unit), 2 layers, T=512 B=64 D=256 H=512.
// fp16 WMMA (v_wmma_f32_16x16x32_f16) with f32 accumulate everywhere.
// ---------------------------------------------------------------------------

#define T_LEN 512
#define BATCH 64
#define D_IN  256
#define HID   512
#define G2    1024   // 2*HID

typedef _Float16 v16h __attribute__((ext_vector_type(16)));
typedef _Float16 v8h  __attribute__((ext_vector_type(8)));
typedef float    v8f  __attribute__((ext_vector_type(8)));

__device__ __forceinline__ v16h cat16(v8h lo, v8h hi) {
  return __builtin_shufflevector(lo, hi, 0,1,2,3,4,5,6,7,8,9,10,11,12,13,14,15);
}

__device__ __forceinline__ v8f wmma_f16(v16h a, v16h b, v8f c) {
  // D = A(16x32 f16) * B(32x16 f16) + C(16x16 f32)
  return __builtin_amdgcn_wmma_f32_16x16x32_f16(
      /*neg_a=*/false, a, /*neg_b=*/false, b,
      /*c_mod=*/(short)0, c, /*reuse_a=*/false, /*reuse_b=*/false);
}

// A-fragment (16x32 f16) from row-major f16 [rows][ldk]:
//   lane L holds row m0+(L&15); halves j<8 -> K = kbase+off+j, j>=8 -> K = kbase+16+off+(j-8)
//   off = 0 for lanes 0-15, 8 for lanes 16-31.  Two contiguous 16B loads.
__device__ __forceinline__ v16h load_a_f16(const _Float16* __restrict__ base,
                                           int row, int ldk, int kbase, int off) {
  const _Float16* p = base + (size_t)row * ldk + kbase + off;
  return cat16(*(const v8h*)p, *(const v8h*)(p + 16));
}

// B-fragment: packed layout, each lane's 16 halves contiguous.
__device__ __forceinline__ v16h load_b_packed(const _Float16* __restrict__ p) {
  return cat16(*(const v8h*)p, *(const v8h*)(p + 8));
}

__device__ __forceinline__ float sigmoidf_(float x) {
  return 1.0f / (1.0f + expf(-x));
}

// ---------------------------------------------------------------------------
// f32 -> f16 elementwise convert (for projection A operands)
// ---------------------------------------------------------------------------
__global__ void conv_f16(const float* __restrict__ src, _Float16* __restrict__ dst, int n) {
  int i = blockIdx.x * blockDim.x + threadIdx.x;
  if (i < n) dst[i] = (_Float16)src[i];
}

// ---------------------------------------------------------------------------
// Weight pack: f32 W[K][1024] row-major -> f16 B-fragments.
// packed[((ntile*KB + kblk)*32 + lane)*16 + j] = W[kblk*32 + 16*(j/8) + 8*(lane/16) + j%8][ntile*16 + lane%16]
// (B-matrix VGPR layout: lane L holds column N=L%16; half j -> K = 16*(j/8) + 8*(L/16) + j%8 within k-block)
// ---------------------------------------------------------------------------
__global__ void pack_weight(const float* __restrict__ W, _Float16* __restrict__ P, int K) {
  int tid = blockIdx.x * blockDim.x + threadIdx.x;
  int total = K * G2;
  if (tid >= total) return;
  int j    = tid & 15;
  int lane = (tid >> 4) & 31;
  int rest = tid >> 9;
  int KB   = K >> 5;
  int kblk  = rest % KB;
  int ntile = rest / KB;
  int srcK = kblk * 32 + ((j >> 3) << 4) + ((lane >> 4) << 3) + (j & 7);
  int srcN = ntile * 16 + (lane & 15);
  P[tid] = (_Float16)W[(size_t)srcK * G2 + srcN];
}

// ---------------------------------------------------------------------------
// Input projection: XW[32768][1024] = A[32768][K](f16) @ Wp(K x 1024, packed f16)
// One wave computes 1 M-tile x 4 N-tiles (A fragment reused 4x).
// grid = (2048/8, 64/4), block = 256 (8 waves).
// ---------------------------------------------------------------------------
__global__ __launch_bounds__(256)
void proj_gemm(const _Float16* __restrict__ A, const _Float16* __restrict__ Wp,
               float* __restrict__ XW, int K) {
  const int lane  = threadIdx.x & 31;
  const int wave  = threadIdx.x >> 5;
  const int mtile = blockIdx.x * 8 + wave;
  const int ng0   = blockIdx.y * 4;          // first of 4 consecutive N-tiles
  const int KB    = K >> 5;
  const int row   = mtile * 16 + (lane & 15);
  const int off   = (lane < 16) ? 0 : 8;

  v8f acc[4] = {{}, {}, {}, {}};
  for (int kblk = 0; kblk < KB; ++kblk) {
    v16h a = load_a_f16(A, row, K, kblk * 32, off);
#pragma unroll
    for (int q = 0; q < 4; ++q) {
      const _Float16* pb = Wp + ((size_t)(ng0 + q) * KB + kblk) * 512 + lane * 16;
      acc[q] = wmma_f16(a, load_b_packed(pb), acc[q]);
    }
  }
  // C/D layout: VGPR r, lane L -> M = m0 + r + 8*(L/16), N = n0 + (L%16)
#pragma unroll
  for (int q = 0; q < 4; ++q) {
    float* outp = XW + (size_t)(mtile * 16 + 8 * (lane >> 4)) * G2 + (ng0 + q) * 16 + (lane & 15);
#pragma unroll
    for (int r = 0; r < 8; ++r) outp[(size_t)r * G2] = acc[q][r];
  }
}

// ---------------------------------------------------------------------------
// Grid-wide barrier (persistent recurrence kernel), 32 blocks.
// ---------------------------------------------------------------------------
__device__ void grid_sync(unsigned* __restrict__ bar, unsigned nb) {
  __syncthreads();
  if (threadIdx.x == 0) {
    __threadfence();
    unsigned* cnt = bar;
    unsigned* gen = bar + 1;
    unsigned g = __hip_atomic_load(gen, __ATOMIC_RELAXED, __HIP_MEMORY_SCOPE_AGENT);
    unsigned arrived = __hip_atomic_fetch_add(cnt, 1u, __ATOMIC_ACQ_REL, __HIP_MEMORY_SCOPE_AGENT);
    if (arrived == nb - 1) {
      __hip_atomic_store(cnt, 0u, __ATOMIC_RELAXED, __HIP_MEMORY_SCOPE_AGENT);
      __hip_atomic_fetch_add(gen, 1u, __ATOMIC_RELEASE, __HIP_MEMORY_SCOPE_AGENT);
    } else {
      while (__hip_atomic_load(gen, __ATOMIC_ACQUIRE, __HIP_MEMORY_SCOPE_AGENT) == g) {
        __builtin_amdgcn_s_sleep(1);
      }
    }
    __threadfence();
  }
  __syncthreads();
}

// ---------------------------------------------------------------------------
// Persistent recurrence kernel.  32 blocks x 256 threads.
// Block b owns gate columns [16b,16b+16) and [512+16b,512+16b+16).
// Per step: 8 waves x 2 tiles (16x16, K=512 -> 16 WMMAs each) = 256 WMMAs/block,
// LDS-stage pre-activations, elementwise sigmoid/tanh update of the block's
// own h/c columns, fp16 state double-buffered, one grid barrier per step.
// Step's gate inputs (XW) are prefetched (global_prefetch_b8) before the GEMM
// so the WMMA phase hides their HBM/L2 latency.
// ---------------------------------------------------------------------------
__global__ __launch_bounds__(256)
void rgu_recur(const float* __restrict__ XW,          // [T*B][1024] input proj
               float* __restrict__ seqF32,            // [T*B][512] h sequence out (may be null)
               _Float16* __restrict__ seqF16,         // [T*B][512] h sequence out (may be null)
               const _Float16* __restrict__ WcgP,     // packed (512x1024)
               const _Float16* __restrict__ WhgP,     // packed (512x1024)
               const float* __restrict__ bcg, const float* __restrict__ bhg,
               float* __restrict__ hF32, float* __restrict__ cF32,       // [64*512]
               _Float16* __restrict__ hF16, _Float16* __restrict__ cF16, // [2][64*512]
               unsigned* __restrict__ barrier,
               float* __restrict__ finalH, float* __restrict__ finalC) {
  __shared__ float sCg[2][64][16];   // c@Wcg pre-act:  [p][m][j]  p=0: col n, p=1: col n+512
  __shared__ float sHg[2][64][16];   // h@Whg pre-act

  const int lane = threadIdx.x & 31;
  const int wave = threadIdx.x >> 5;
  const int mt   = wave & 3;          // row tile 0..3  (rows 16*mt ..)
  const int p    = wave >> 2;         // 0: gate col n, 1: gate col n+512
  const int blk  = blockIdx.x;        // 0..31
  const int ntile = blk + p * 32;     // column tile in [0,64)
  const int off  = (lane < 16) ? 0 : 8;
  const int arow = mt * 16 + (lane & 15);

  // per-thread elementwise constants: column j within the block's slab
  const int j = threadIdx.x & 15;
  const int n = blk * 16 + j;
  const float bc0 = bcg[n], bc1 = bcg[n + 512];
  const float bh0 = bhg[n], bh1 = bhg[n + 512];

  const _Float16* bcgp = WcgP + (size_t)(ntile * 16) * 512;  // KB = 16
  const _Float16* bhgp = WhgP + (size_t)(ntile * 16) * 512;

  for (int t = 0; t < T_LEN; ++t) {
    const _Float16* hcur = hF16 + (size_t)(t & 1) * (BATCH * HID);
    const _Float16* ccur = cF16 + (size_t)(t & 1) * (BATCH * HID);
    _Float16* hnxt = hF16 + (size_t)((t + 1) & 1) * (BATCH * HID);
    _Float16* cnxt = cF16 + (size_t)((t + 1) & 1) * (BATCH * HID);

    // ---- prefetch this step's gate inputs; GEMM phase hides the latency ----
    const float* xp = XW + (size_t)(t * BATCH) * G2 + n;
#pragma unroll
    for (int i = 0; i < 4; ++i) {
      int m = (threadIdx.x >> 4) + 16 * i;
      __builtin_prefetch(xp + (size_t)m * G2, 0, 3);
      __builtin_prefetch(xp + (size_t)m * G2 + 512, 0, 3);
    }

    // ---- GEMM phase: accCg = (c @ Wcg) tile, accHg = (h @ Whg) tile ----
    v8f accCg = {}, accHg = {};
#pragma unroll 4
    for (int kblk = 0; kblk < 16; ++kblk) {
      v16h aC = load_a_f16(ccur, arow, HID, kblk * 32, off);
      v16h aH = load_a_f16(hcur, arow, HID, kblk * 32, off);
      v16h bC = load_b_packed(bcgp + kblk * 512 + lane * 16);
      v16h bH = load_b_packed(bhgp + kblk * 512 + lane * 16);
      accCg = wmma_f16(aC, bC, accCg);
      accHg = wmma_f16(aH, bH, accHg);
    }
#pragma unroll
    for (int r = 0; r < 8; ++r) {
      int row = mt * 16 + r + 8 * (lane >> 4);
      sCg[p][row][lane & 15] = accCg[r];
      sHg[p][row][lane & 15] = accHg[r];
    }
    __syncthreads();

    // ---- elementwise gate/update phase: 4 elements per thread ----
#pragma unroll
    for (int i = 0; i < 4; ++i) {
      int m = (threadIdx.x >> 4) + 16 * i;
      float xh = XW[(size_t)(t * BATCH + m) * G2 + n];
      float xc = XW[(size_t)(t * BATCH + m) * G2 + 512 + n];
      float hOld = hF32[m * HID + n];
      float cOld = cF32[m * HID + n];
      float gh_in  = sigmoidf_(sCg[0][m][j] + bc0);
      float gc_rec = sigmoidf_(sCg[1][m][j] + bc1);
      float gh_rec = sigmoidf_(sHg[0][m][j] + bh0);
      float gc_in  = sigmoidf_(sHg[1][m][j] + bh1);
      float hn = tanhf((1.0f - gh_in) * xh + (1.0f - gh_rec) * hOld);
      float cn = tanhf((1.0f - gc_in) * xc + (1.0f - gc_rec) * cOld);
      hF32[m * HID + n] = hn;
      cF32[m * HID + n] = cn;
      hnxt[m * HID + n] = (_Float16)hn;
      cnxt[m * HID + n] = (_Float16)cn;
      size_t sidx = (size_t)(t * BATCH + m) * HID + n;
      if (seqF32 != nullptr) seqF32[sidx] = hn;
      if (seqF16 != nullptr) seqF16[sidx] = (_Float16)hn;
      if (finalH != nullptr && t == T_LEN - 1) {
        finalH[m * HID + n] = hn;
        finalC[m * HID + n] = cn;
      }
    }

    grid_sync(barrier, gridDim.x);   // state writes visible before next step's reads
  }
}

// ---------------------------------------------------------------------------
// host-side launch
// ---------------------------------------------------------------------------
extern "C" void kernel_launch(void* const* d_in, const int* in_sizes, int n_in,
                              void* d_out, int out_size, void* d_ws, size_t ws_size,
                              hipStream_t stream) {
  const float* x    = (const float*)d_in[0];
  const float* Wx0  = (const float*)d_in[1];
  const float* Wcg0 = (const float*)d_in[2];
  const float* Whg0 = (const float*)d_in[3];
  const float* bcg0 = (const float*)d_in[4];
  const float* bhg0 = (const float*)d_in[5];
  const float* Wx1  = (const float*)d_in[6];
  const float* Wcg1 = (const float*)d_in[7];
  const float* Whg1 = (const float*)d_in[8];
  const float* bcg1 = (const float*)d_in[9];
  const float* bhg1 = (const float*)d_in[10];
  float* out = (float*)d_out;

  // workspace layout
  char* ws = (char*)d_ws;
  size_t off = 0;
  auto alloc = [&](size_t bytes) {
    char* p = ws + off;
    off += (bytes + 255) & ~(size_t)255;
    return p;
  };
  unsigned*  barrier = (unsigned*)alloc(256);
  _Float16*  hF16  = (_Float16*)alloc((size_t)2 * BATCH * HID * sizeof(_Float16));
  _Float16*  cF16  = (_Float16*)alloc((size_t)2 * BATCH * HID * sizeof(_Float16));
  float*     hF32  = (float*)alloc((size_t)BATCH * HID * sizeof(float));
  float*     cF32  = (float*)alloc((size_t)BATCH * HID * sizeof(float));
  _Float16*  WxP0  = (_Float16*)alloc((size_t)D_IN * G2 * sizeof(_Float16));
  _Float16*  WcgP0 = (_Float16*)alloc((size_t)HID * G2 * sizeof(_Float16));
  _Float16*  WhgP0 = (_Float16*)alloc((size_t)HID * G2 * sizeof(_Float16));
  _Float16*  WxP1  = (_Float16*)alloc((size_t)HID * G2 * sizeof(_Float16));
  _Float16*  WcgP1 = (_Float16*)alloc((size_t)HID * G2 * sizeof(_Float16));
  _Float16*  WhgP1 = (_Float16*)alloc((size_t)HID * G2 * sizeof(_Float16));
  _Float16*  xF16  = (_Float16*)alloc((size_t)T_LEN * BATCH * D_IN * sizeof(_Float16)); // 16MB
  _Float16*  seqH  = (_Float16*)alloc((size_t)T_LEN * BATCH * HID * sizeof(_Float16));  // 32MB
  float*     XW    = (float*)alloc((size_t)T_LEN * BATCH * G2 * sizeof(float));         // 128MB

  // pack all weights to WMMA-fragment-native f16 layout; convert x to f16
  pack_weight<<<(D_IN * G2 + 255) / 256, 256, 0, stream>>>(Wx0,  WxP0,  D_IN);
  pack_weight<<<(HID * G2 + 255) / 256, 256, 0, stream>>>(Wcg0, WcgP0, HID);
  pack_weight<<<(HID * G2 + 255) / 256, 256, 0, stream>>>(Whg0, WhgP0, HID);
  pack_weight<<<(HID * G2 + 255) / 256, 256, 0, stream>>>(Wx1,  WxP1,  HID);
  pack_weight<<<(HID * G2 + 255) / 256, 256, 0, stream>>>(Wcg1, WcgP1, HID);
  pack_weight<<<(HID * G2 + 255) / 256, 256, 0, stream>>>(Whg1, WhgP1, HID);
  {
    int n = T_LEN * BATCH * D_IN;
    conv_f16<<<(n + 255) / 256, 256, 0, stream>>>(x, xF16, n);
  }

  const size_t stateF16Bytes = (size_t)2 * BATCH * HID * sizeof(_Float16);
  const size_t stateF32Bytes = (size_t)BATCH * HID * sizeof(float);
  const size_t seqElems      = (size_t)T_LEN * BATCH * HID;

  // ---- layer 0 ----
  hipMemsetAsync(barrier, 0, 256, stream);
  hipMemsetAsync(hF16, 0, stateF16Bytes, stream);
  hipMemsetAsync(cF16, 0, stateF16Bytes, stream);
  hipMemsetAsync(hF32, 0, stateF32Bytes, stream);
  hipMemsetAsync(cF32, 0, stateF32Bytes, stream);
  proj_gemm<<<dim3(256, 16), 256, 0, stream>>>(xF16, WxP0, XW, D_IN);
  rgu_recur<<<32, 256, 0, stream>>>(XW, nullptr, seqH, WcgP0, WhgP0, bcg0, bhg0,
                                    hF32, cF32, hF16, cF16, barrier,
                                    nullptr, nullptr);

  // ---- layer 1 ----
  hipMemsetAsync(barrier, 0, 256, stream);
  hipMemsetAsync(hF16, 0, stateF16Bytes, stream);
  hipMemsetAsync(cF16, 0, stateF16Bytes, stream);
  hipMemsetAsync(hF32, 0, stateF32Bytes, stream);
  hipMemsetAsync(cF32, 0, stateF32Bytes, stream);
  proj_gemm<<<dim3(256, 16), 256, 0, stream>>>(seqH, WxP1, XW, HID);
  rgu_recur<<<32, 256, 0, stream>>>(XW, out, nullptr, WcgP1, WhgP1, bcg1, bhg1,
                                    hF32, cF32, hF16, cF16, barrier,
                                    out + seqElems,
                                    out + seqElems + (size_t)BATCH * HID);
}
